// MultiHeadAttention_8856222564798
// MI455X (gfx1250) — matrix operand order repaired and checked by reading, not compiled, not run on hardware
//
#include <hip/hip_runtime.h>

// ---------------------------------------------------------------------------
// MHA (B=2, S=4096, D=768, H=12, Hd=64) via bf16 WMMA flash attention.
// All GEMMs use v_wmma_f32_16x16x32_bf16 (wave32, 16x16x32 tiles, fp32 accum).
// 32-row M-tiles per wave: each B-fragment feeds two WMMAs (halved operand
// fetch per FLOP vs 16-row tiles).
// ---------------------------------------------------------------------------

#define DIM      768
#define NHEADS   12
#define HEADDIM  64
#define BATCH    2
#define SEQ      4096
#define ROWS     (BATCH * SEQ)          // 8192 flat token rows

typedef __attribute__((ext_vector_type(16))) __bf16 bf16x16;
typedef __attribute__((ext_vector_type(8)))  float  floatx8;

union Frag16 {
    unsigned int u[8];
    bf16x16      v;
};

static __device__ __forceinline__ unsigned short f2bf(float f) {
    unsigned int u = __builtin_bit_cast(unsigned int, f);
    unsigned int r = u + 0x7FFFu + ((u >> 16) & 1u);   // round-to-nearest-even
    return (unsigned short)(r >> 16);
}

// A-fragment, 16-bit elements, 16(M) x 32(K). ISA 7.12.2 layout:
// lane l: m = l&15, K-half = l>>4; VGPR v packs K = kbase, kbase+1.
static __device__ __forceinline__ bf16x16 load_a16(const unsigned short* base, int ld) {
    const int l  = threadIdx.x & 31;
    const int m  = l & 15;
    const int kh = l >> 4;
    Frag16 f;
#pragma unroll
    for (int v = 0; v < 8; ++v) {
        const int k = ((v & 4) ? 16 : 0) + kh * 8 + (v & 3) * 2;
        f.u[v] = *(const unsigned int*)(base + m * ld + k);
    }
    return f.v;
}

// B-fragment, 16-bit elements, 32(K) x 16(N), source stored n-major [n][ld]
// with K contiguous. lane l: n = l&15; K = (v>=4?16:0) + 8*(l>>4) + 2*(v&3) (+1 packed).
static __device__ __forceinline__ bf16x16 load_b16(const unsigned short* base, int ld) {
    const int l  = threadIdx.x & 31;
    const int n  = l & 15;
    const int kh = l >> 4;
    Frag16 f;
#pragma unroll
    for (int v = 0; v < 8; ++v) {
        const int k = ((v & 4) ? 16 : 0) + kh * 8 + (v & 3) * 2;
        f.u[v] = *(const unsigned int*)(base + n * ld + k);
    }
    return f.v;
}

static __device__ __forceinline__ floatx8 wmma_bf16(bf16x16 a, bf16x16 b, floatx8 c) {
    return __builtin_amdgcn_wmma_f32_16x16x32_bf16(
        /*neg_a=*/false, a, /*neg_b=*/false, b,
        /*c_mod=*/(short)0, c, /*reuse_a=*/false, /*reuse_b=*/false);
}

// ------------------------------ prep kernels -------------------------------

__global__ void cvt_f32_bf16_kernel(const float* __restrict__ src,
                                    unsigned short* __restrict__ dst, int n) {
    int i = blockIdx.x * 256 + threadIdx.x;
    if (i < n) dst[i] = f2bf(src[i]);
}

// W[k][n] (fp32, y = x@W) -> Wt[n][k] (bf16) so B-fragment K-pairs are contiguous
__global__ void transpose_w_kernel(const float* __restrict__ W,
                                   unsigned short* __restrict__ Wt) {
    int i = blockIdx.x * 256 + threadIdx.x;
    if (i < DIM * DIM) {
        int n = i / DIM, k = i % DIM;
        Wt[n * DIM + k] = f2bf(W[k * DIM + n]);
    }
}

// ------------------------- QKV projection (bf16) ---------------------------
// grid (ROWS/32, NHEADS); one wave computes a 32x64 tile of x@W + b.
// mode 0/1: store [B,H,S,64]; mode 2: store transposed [B,H,64,S].
__global__ void proj32_kernel(const unsigned short* __restrict__ xb,
                              const unsigned short* __restrict__ Wt,
                              const float* __restrict__ bias,
                              unsigned short* __restrict__ out, int mode) {
    const int r = blockIdx.x, h = blockIdx.y;
    const int row0 = r * 32;
    const int b    = row0 >> 12;        // 4096 rows per batch
    const int s0   = row0 & (SEQ - 1);
    const int l    = threadIdx.x & 31;
    const int nl   = l & 15, half = l >> 4;

    floatx8 accA[4] = {}, accB[4] = {};
    for (int kk = 0; kk < DIM; kk += 32) {
        bf16x16 a0 = load_a16(xb + row0 * DIM + kk, DIM);
        bf16x16 a1 = load_a16(xb + (row0 + 16) * DIM + kk, DIM);
#pragma unroll
        for (int nt = 0; nt < 4; ++nt) {
            bf16x16 bm = load_b16(Wt + (h * HEADDIM + nt * 16) * DIM + kk, DIM);
            accA[nt] = wmma_bf16(a0, bm, accA[nt]);
            accB[nt] = wmma_bf16(a1, bm, accB[nt]);
        }
    }

    const int bh = b * NHEADS + h;
#pragma unroll
    for (int nt = 0; nt < 4; ++nt) {
        const int hd = nt * 16 + nl;
        const float bv = bias[h * HEADDIM + hd];
#pragma unroll
        for (int v = 0; v < 8; ++v) {
            const int m = v + 8 * half;
            const unsigned short yA = f2bf(accA[nt][v] + bv);
            const unsigned short yB = f2bf(accB[nt][v] + bv);
            if (mode == 2) {  // V transposed: [B,H,64,S]
                out[(bh * HEADDIM + hd) * SEQ + s0 + m]      = yA;
                out[(bh * HEADDIM + hd) * SEQ + s0 + 16 + m] = yB;
            } else {          // Q/K: [B,H,S,64]
                out[(bh * SEQ + s0 + m) * HEADDIM + hd]      = yA;
                out[(bh * SEQ + s0 + 16 + m) * HEADDIM + hd] = yB;
            }
        }
    }
}

// -------------------- flash attention online-softmax step ------------------
// One 16-query x 32-key score tile in C-layout (s0 = keys 0..15, s1 = 16..31).
// Updates running max/sum, rescales O accumulators, writes exp(P) as bf16 to
// the LDS staging tile at row offset `rowoff`.
static __device__ __forceinline__ void softmax_tile(
        floatx8 s0, floatx8 s1, float* rmax, float* rsum, floatx8* o,
        unsigned short* plds, int rowoff, int nl, int half) {
    const float scale = 0.125f;   // 1/sqrt(64)
#pragma unroll
    for (int v = 0; v < 8; ++v) {
        float a  = s0[v] * scale;
        float bb = s1[v] * scale;
        float mv = fmaxf(a, bb);
        mv = fmaxf(mv, __shfl_xor(mv, 1, 32));
        mv = fmaxf(mv, __shfl_xor(mv, 2, 32));
        mv = fmaxf(mv, __shfl_xor(mv, 4, 32));
        mv = fmaxf(mv, __shfl_xor(mv, 8, 32));
        const float nm = fmaxf(rmax[v], mv);
        const float c  = __expf(rmax[v] - nm);
        rmax[v] = nm;
        const float p0 = __expf(a - nm);
        const float p1 = __expf(bb - nm);
        float ls = p0 + p1;
        ls += __shfl_xor(ls, 1, 32);
        ls += __shfl_xor(ls, 2, 32);
        ls += __shfl_xor(ls, 4, 32);
        ls += __shfl_xor(ls, 8, 32);
        rsum[v] = rsum[v] * c + ls;
        o[0][v] *= c; o[1][v] *= c; o[2][v] *= c; o[3][v] *= c;
        const int m = rowoff + v + 8 * half;     // C-layout -> row-major P tile
        plds[m * 32 + nl]      = f2bf(p0);
        plds[m * 32 + 16 + nl] = f2bf(p1);
    }
}

// ------------------------ flash attention (bf16) ---------------------------
// grid (SEQ/32, NHEADS, BATCH); one wave owns a 32-query tile, streams 32-key
// blocks: 8 WMMAs QK^T, fp32 online softmax, LDS relayout of P, 8 WMMAs P*V.
// Each K/V B-fragment feeds two WMMAs.
__global__ void flash_kernel(const unsigned short* __restrict__ Q,
                             const unsigned short* __restrict__ K,
                             const unsigned short* __restrict__ Vt,
                             unsigned short* __restrict__ attn) {
    __shared__ unsigned short plds[32 * 32];

    const int qt = blockIdx.x, h = blockIdx.y, b = blockIdx.z;
    const int l  = threadIdx.x & 31;
    const int nl = l & 15, half = l >> 4;
    const int bh = b * NHEADS + h;

    const unsigned short* Qb = Q + (bh * SEQ + qt * 32) * HEADDIM;
    const bf16x16 qA0 = load_a16(Qb,                HEADDIM);
    const bf16x16 qA1 = load_a16(Qb + 32,           HEADDIM);
    const bf16x16 qB0 = load_a16(Qb + 16 * HEADDIM,      HEADDIM);
    const bf16x16 qB1 = load_a16(Qb + 16 * HEADDIM + 32, HEADDIM);

    floatx8 oA[4] = {}, oB[4] = {};
    float rmaxA[8], rsumA[8], rmaxB[8], rsumB[8];
#pragma unroll
    for (int v = 0; v < 8; ++v) {
        rmaxA[v] = -3.4e38f; rsumA[v] = 0.0f;
        rmaxB[v] = -3.4e38f; rsumB[v] = 0.0f;
    }

    for (int kb = 0; kb < SEQ; kb += 32) {
        const unsigned short* Kb = K + (bh * SEQ + kb) * HEADDIM;
        if (kb + 32 < SEQ)        // pull next 4KB K tile toward the WGP
            __builtin_prefetch(Kb + 32 * HEADDIM + l * 64, 0, 3);

        const bf16x16 k0 = load_b16(Kb,                     HEADDIM);
        const bf16x16 k1 = load_b16(Kb + 32,                HEADDIM);
        const bf16x16 k2 = load_b16(Kb + 16 * HEADDIM,      HEADDIM);
        const bf16x16 k3 = load_b16(Kb + 16 * HEADDIM + 32, HEADDIM);

        floatx8 s0A = {}, s1A = {}, s0B = {}, s1B = {};
        s0A = wmma_bf16(qA0, k0, s0A); s0A = wmma_bf16(qA1, k1, s0A);
        s1A = wmma_bf16(qA0, k2, s1A); s1A = wmma_bf16(qA1, k3, s1A);
        s0B = wmma_bf16(qB0, k0, s0B); s0B = wmma_bf16(qB1, k1, s0B);
        s1B = wmma_bf16(qB0, k2, s1B); s1B = wmma_bf16(qB1, k3, s1B);

        __syncthreads();
        softmax_tile(s0A, s1A, rmaxA, rsumA, oA, plds, 0,  nl, half);
        softmax_tile(s0B, s1B, rmaxB, rsumB, oB, plds, 16, nl, half);
        __syncthreads();

        const bf16x16 pA = load_a16(plds,           32);
        const bf16x16 pB = load_a16(plds + 16 * 32, 32);
        const unsigned short* Vb = Vt + (bh * HEADDIM) * SEQ + kb;
#pragma unroll
        for (int nt = 0; nt < 4; ++nt) {
            const bf16x16 vb = load_b16(Vb + nt * 16 * SEQ, SEQ);
            oA[nt] = wmma_bf16(pA, vb, oA[nt]);
            oB[nt] = wmma_bf16(pB, vb, oB[nt]);
        }
    }

    unsigned short* ab = attn + (b * SEQ + qt * 32) * DIM + h * HEADDIM;
#pragma unroll
    for (int v = 0; v < 8; ++v) {
        const int m = v + 8 * half;
        const float invA = 1.0f / rsumA[v];
        const float invB = 1.0f / rsumB[v];
#pragma unroll
        for (int nt = 0; nt < 4; ++nt) {
            ab[m * DIM + nt * 16 + nl]        = f2bf(oA[nt][v] * invA);
            ab[(16 + m) * DIM + nt * 16 + nl] = f2bf(oB[nt][v] * invB);
        }
    }
}

// ------------------------- output projection (fp32 out) --------------------
__global__ void outproj_kernel(const unsigned short* __restrict__ attnb,
                               const unsigned short* __restrict__ Wt,
                               const float* __restrict__ bias,
                               float* __restrict__ out) {
    const int r = blockIdx.x, h = blockIdx.y;
    const int row0 = r * 32;
    const int l  = threadIdx.x & 31;
    const int nl = l & 15, half = l >> 4;

    floatx8 accA[4] = {}, accB[4] = {};
    for (int kk = 0; kk < DIM; kk += 32) {
        bf16x16 a0 = load_a16(attnb + row0 * DIM + kk, DIM);
        bf16x16 a1 = load_a16(attnb + (row0 + 16) * DIM + kk, DIM);
#pragma unroll
        for (int nt = 0; nt < 4; ++nt) {
            bf16x16 bm = load_b16(Wt + (h * HEADDIM + nt * 16) * DIM + kk, DIM);
            accA[nt] = wmma_bf16(a0, bm, accA[nt]);
            accB[nt] = wmma_bf16(a1, bm, accB[nt]);
        }
    }
#pragma unroll
    for (int nt = 0; nt < 4; ++nt) {
        const int ng = h * HEADDIM + nt * 16 + nl;
        const float bv = bias[ng];
#pragma unroll
        for (int v = 0; v < 8; ++v) {
            const int m = v + 8 * half;
            out[(row0 + m) * DIM + ng]      = accA[nt][v] + bv;
            out[(row0 + 16 + m) * DIM + ng] = accB[nt][v] + bv;
        }
    }
}

// ------------------------------- launcher ----------------------------------
extern "C" void kernel_launch(void* const* d_in, const int* in_sizes, int n_in,
                              void* d_out, int out_size, void* d_ws, size_t ws_size,
                              hipStream_t stream) {
    const float* x  = (const float*)d_in[0];
    const float* Wq = (const float*)d_in[1];
    const float* bq = (const float*)d_in[2];
    const float* Wk = (const float*)d_in[3];
    const float* bk = (const float*)d_in[4];
    const float* Wv = (const float*)d_in[5];
    const float* bv = (const float*)d_in[6];
    const float* Wo = (const float*)d_in[7];
    const float* bo = (const float*)d_in[8];
    float* out = (float*)d_out;

    char* ws = (char*)d_ws;
    const size_t XB   = (size_t)ROWS * DIM * 2;                     // 12.6 MB
    const size_t WT   = (size_t)DIM * DIM * 2;                      // 1.18 MB
    const size_t QKV  = (size_t)BATCH * NHEADS * SEQ * HEADDIM * 2; // 12.6 MB
    unsigned short* xb   = (unsigned short*)(ws);
    unsigned short* wtq  = (unsigned short*)(ws + XB);
    unsigned short* wtk  = (unsigned short*)(ws + XB + WT);
    unsigned short* wtv  = (unsigned short*)(ws + XB + 2 * WT);
    unsigned short* wto  = (unsigned short*)(ws + XB + 3 * WT);
    unsigned short* Qb   = (unsigned short*)(ws + XB + 4 * WT);
    unsigned short* Kb   = (unsigned short*)(ws + XB + 4 * WT + QKV);
    unsigned short* Vtb  = (unsigned short*)(ws + XB + 4 * WT + 2 * QKV);
    unsigned short* attn = (unsigned short*)(ws + XB + 4 * WT + 3 * QKV);
    // total ~67.6 MB of workspace

    // 1) convert x -> bf16
    {
        int n = ROWS * DIM;
        cvt_f32_bf16_kernel<<<(n + 255) / 256, 256, 0, stream>>>(x, xb, n);
    }
    // 2) transpose+convert weights
    {
        int n = DIM * DIM, g = (n + 255) / 256;
        transpose_w_kernel<<<g, 256, 0, stream>>>(Wq, wtq);
        transpose_w_kernel<<<g, 256, 0, stream>>>(Wk, wtk);
        transpose_w_kernel<<<g, 256, 0, stream>>>(Wv, wtv);
        transpose_w_kernel<<<g, 256, 0, stream>>>(Wo, wto);
    }
    // 3) Q/K/V projections (32x64 tiles per wave)
    {
        dim3 grid(ROWS / 32, NHEADS);
        proj32_kernel<<<grid, 32, 0, stream>>>(xb, wtq, bq, Qb, 0);
        proj32_kernel<<<grid, 32, 0, stream>>>(xb, wtk, bk, Kb, 1);
        proj32_kernel<<<grid, 32, 0, stream>>>(xb, wtv, bv, Vtb, 2);
    }
    // 4) flash attention (32-query tiles per wave)
    {
        dim3 grid(SEQ / 32, NHEADS, BATCH);
        flash_kernel<<<grid, 32, 0, stream>>>(Qb, Kb, Vtb, attn);
    }
    // 5) output projection (32x64 tiles per wave)
    {
        dim3 grid(ROWS / 32, NHEADS);
        outproj_kernel<<<grid, 32, 0, stream>>>(attn, wto, bo, out);
    }
}
